// graph_network_18906446037564
// MI455X (gfx1250) — compile-verified, matrix operand order checked by reading, not run on hardware
//
#include <hip/hip_runtime.h>

typedef __bf16 bf16_t;
typedef bf16_t v16bf __attribute__((ext_vector_type(16)));
typedef bf16_t v8bf  __attribute__((ext_vector_type(8)));
typedef float  v8f   __attribute__((ext_vector_type(8)));

#define WPB 8  // waves per 256-thread block (wave32)

__device__ __forceinline__ v8f zero8() {
  v8f z = {0.f, 0.f, 0.f, 0.f, 0.f, 0.f, 0.f, 0.f};
  return z;
}

__device__ __forceinline__ v8f wmma_bf16(v16bf a, v16bf b, v8f c) {
  // D = A(16x32 bf16) x B(32x16 bf16) + C(16x16 f32)
  return __builtin_amdgcn_wmma_f32_16x16x32_bf16(false, a, false, b, (short)0, c, false, false);
}

// 16-bit operand fragment per CDNA5 ISA layout:
// lane L -> row (L&15); k-base ((L>>4)*8); elems [0..7]=K kb..kb+7, [8..15]=K kb+16..kb+23
__device__ __forceinline__ v16bf load_frag(const bf16_t* base, int stride, int lane) {
  const int row = lane & 15;
  const int kb  = (lane >> 4) << 3;
  const bf16_t* p = base + row * stride + kb;
  v8bf lo = *(const v8bf*)(p);
  v8bf hi = *(const v8bf*)(p + 16);
  v16bf f;
#pragma unroll
  for (int i = 0; i < 8; ++i) { f[i] = lo[i]; f[i + 8] = hi[i]; }
  return f;
}

// lrelu(v) = v>=0 ? v : 0.05v  ==  max(v, 0.05v)   (single v_max_f32)
__device__ __forceinline__ float lrelu(float v) { return fmaxf(v, 0.05f * v); }

// Two-layer MLP (K1 = KC1*32 -> 64 -> 64) + LayerNorm on a 16-row tile, per wave.
// xbuf: LDS [16][xs] bf16 inputs. hbuf: LDS [16][64] bf16 scratch.
// Result in fragment layout: outv[nt][r] = element (M=(lane>>4)*8+r, N=nt*16+(lane&15)).
template<int KC1>
__device__ __forceinline__ void mlp_tile(const bf16_t* xbuf, int xs,
                                         bf16_t* hbuf,
                                         const bf16_t* W1t, int w1s, const float* b1,
                                         const bf16_t* W2t, const float* b2,
                                         const float* g, const float* be,
                                         int lane, float outv[4][8]) {
  const int ncol  = lane & 15;
  const int mbase = (lane >> 4) << 3;

  v8f acc[4] = {zero8(), zero8(), zero8(), zero8()};
  // Keep K loop rolled: only 1 A-frag + 4 B-frags live -> low VGPR pressure,
  // high occupancy to hide the gather latency that dominates this kernel.
#pragma unroll 1
  for (int kc = 0; kc < KC1; ++kc) {
    v16bf a = load_frag(xbuf + kc * 32, xs, lane);
#pragma unroll
    for (int nt = 0; nt < 4; ++nt) {
      v16bf b = load_frag(W1t + nt * 16 * w1s + kc * 32, w1s, lane);
      acc[nt] = wmma_bf16(a, b, acc[nt]);
    }
  }
  // bias + lrelu, stage h (bf16) for second GEMM
#pragma unroll
  for (int nt = 0; nt < 4; ++nt) {
    const int N = nt * 16 + ncol;
    const float bb = b1[N];
#pragma unroll
    for (int r = 0; r < 8; ++r)
      hbuf[(mbase + r) * 64 + N] = (bf16_t)lrelu(acc[nt][r] + bb);
  }

  v8f acc2[4] = {zero8(), zero8(), zero8(), zero8()};
#pragma unroll 1
  for (int kc = 0; kc < 2; ++kc) {
    v16bf a = load_frag(hbuf + kc * 32, 64, lane);
#pragma unroll
    for (int nt = 0; nt < 4; ++nt) {
      v16bf b = load_frag(W2t + nt * 16 * 64 + kc * 32, 64, lane);
      acc2[nt] = wmma_bf16(a, b, acc2[nt]);
    }
  }
#pragma unroll
  for (int nt = 0; nt < 4; ++nt) {
    const int N = nt * 16 + ncol;
    const float bb = b2[N];
#pragma unroll
    for (int r = 0; r < 8; ++r) outv[nt][r] = lrelu(acc2[nt][r] + bb);
  }

  // LayerNorm: row M = mbase+r lives across 16 lanes of this half-wave, 4 values/lane.
#pragma unroll
  for (int r = 0; r < 8; ++r) {
    float s = outv[0][r] + outv[1][r] + outv[2][r] + outv[3][r];
    float q = outv[0][r] * outv[0][r] + outv[1][r] * outv[1][r]
            + outv[2][r] * outv[2][r] + outv[3][r] * outv[3][r];
#pragma unroll
    for (int m = 1; m <= 8; m <<= 1) {
      s += __shfl_xor(s, m, 32);
      q += __shfl_xor(q, m, 32);
    }
    const float mean = s * (1.0f / 64.0f);
    const float var  = q * (1.0f / 64.0f) - mean * mean;
    const float rs   = rsqrtf(var + 1e-5f);
#pragma unroll
    for (int nt = 0; nt < 4; ++nt) {
      const int N = nt * 16 + ncol;
      outv[nt][r] = (outv[nt][r] - mean) * rs * g[N] + be[N];
    }
  }
}

// ---------------- weight prep: W[K][Nc] f32 -> Wt[Np][Kp] bf16 (zero padded) -----------
__global__ void transpose_cvt_kernel(const float* __restrict__ W, bf16_t* __restrict__ Wt,
                                     int K, int Nc, int Kp, int Np) {
  int idx = blockIdx.x * blockDim.x + threadIdx.x;
  if (idx >= Kp * Np) return;
  int nrow = idx / Kp;
  int k    = idx - nrow * Kp;
  float v = (nrow < Nc && k < K) ? W[k * Nc + nrow] : 0.0f;
  Wt[idx] = (bf16_t)v;
}

__global__ void zero_kernel(float* __restrict__ p, int count) {
  int i = blockIdx.x * blockDim.x + threadIdx.x;
  if (i < count) p[i] = 0.0f;
}

// ---------------- encoders: rows x CIN -> MLP -> out rows x 64 -------------------------
template<int CIN>
__global__ void __launch_bounds__(256) encoder_kernel(const float* __restrict__ xin,
    const bf16_t* __restrict__ W1t, const float* __restrict__ b1,
    const bf16_t* __restrict__ W2t, const float* __restrict__ b2,
    const float* __restrict__ g, const float* __restrict__ be,
    float* __restrict__ out, int rows) {
  __shared__ alignas(16) bf16_t sx[WPB * 16 * 32];
  __shared__ alignas(16) bf16_t sh[WPB * 16 * 64];
  const int lane = threadIdx.x & 31;
  const int wv   = threadIdx.x >> 5;
  bf16_t* xbuf = sx + wv * 16 * 32;
  bf16_t* hbuf = sh + wv * 16 * 64;
  const int nw = gridDim.x * WPB;
  const int ntiles = rows >> 4;
  const int ncol  = lane & 15;
  const int mbase = (lane >> 4) << 3;
  for (int tile = blockIdx.x * WPB + wv; tile < ntiles; tile += nw) {
    const int row0 = tile << 4;
    for (int m = 0; m < 16; ++m) {
      float v = (lane < CIN) ? xin[(row0 + m) * CIN + lane] : 0.0f;
      xbuf[m * 32 + lane] = (bf16_t)v;
    }
    float outv[4][8];
    mlp_tile<1>(xbuf, 32, hbuf, W1t, 32, b1, W2t, b2, g, be, lane, outv);
#pragma unroll
    for (int r = 0; r < 8; ++r) {
      const int row = row0 + mbase + r;
#pragma unroll
      for (int nt = 0; nt < 4; ++nt)
        out[row * 64 + nt * 16 + ncol] = outv[nt][r];
    }
  }
}

// ---------------- edge update: e += MLP([e|n[s]|n[r]]); aggr[s] += e -------------------
__global__ void __launch_bounds__(256) edge_update_kernel(float* __restrict__ e,
    const float* __restrict__ n, const int* __restrict__ edge_idx,
    const bf16_t* __restrict__ W1t, const float* __restrict__ b1,
    const bf16_t* __restrict__ W2t, const float* __restrict__ b2,
    const float* __restrict__ g, const float* __restrict__ be,
    float* __restrict__ aggr, int E_) {
  __shared__ alignas(16) bf16_t sx[WPB * 16 * 192];
  __shared__ alignas(16) bf16_t sh[WPB * 16 * 64];
  const int lane = threadIdx.x & 31;
  const int wv   = threadIdx.x >> 5;
  bf16_t* xbuf = sx + wv * 16 * 192;
  bf16_t* hbuf = sh + wv * 16 * 64;
  const int nw = gridDim.x * WPB;
  const int ntiles = E_ >> 4;
  const int ncol  = lane & 15;
  const int mbase = (lane >> 4) << 3;
  for (int tile = blockIdx.x * WPB + wv; tile < ntiles; tile += nw) {
    const int row0 = tile << 4;
    __builtin_prefetch(e + (size_t)row0 * 64, 0, 3);
    for (int m = 0; m < 16; ++m) {
      const int row  = row0 + m;                 // wave-uniform
      const int sidx = edge_idx[row * 2 + 0];
      const int ridx = edge_idx[row * 2 + 1];
      for (int c = lane; c < 192; c += 32) {
        float v;
        if (c < 64)       v = e[row * 64 + c];
        else if (c < 128) v = n[sidx * 64 + (c - 64)];
        else              v = n[ridx * 64 + (c - 128)];
        xbuf[m * 192 + c] = (bf16_t)v;
      }
    }
    float outv[4][8];
    mlp_tile<6>(xbuf, 192, hbuf, W1t, 192, b1, W2t, b2, g, be, lane, outv);
    // residual store + fused segment_sum scatter
#pragma unroll
    for (int r = 0; r < 8; ++r) {
      const int row  = row0 + mbase + r;
      const int sidx = edge_idx[row * 2 + 0];
#pragma unroll
      for (int nt = 0; nt < 4; ++nt) {
        const int N = nt * 16 + ncol;
        float env = e[row * 64 + N] + outv[nt][r];
        e[row * 64 + N] = env;
        atomicAdd(&aggr[sidx * 64 + N], env);
      }
    }
  }
}

// ---------------- node update: n += MLP([n|aggr]) --------------------------------------
__global__ void __launch_bounds__(256) node_update_kernel(float* __restrict__ n,
    const float* __restrict__ aggr,
    const bf16_t* __restrict__ W1t, const float* __restrict__ b1,
    const bf16_t* __restrict__ W2t, const float* __restrict__ b2,
    const float* __restrict__ g, const float* __restrict__ be, int N_) {
  __shared__ alignas(16) bf16_t sx[WPB * 16 * 128];
  __shared__ alignas(16) bf16_t sh[WPB * 16 * 64];
  const int lane = threadIdx.x & 31;
  const int wv   = threadIdx.x >> 5;
  bf16_t* xbuf = sx + wv * 16 * 128;
  bf16_t* hbuf = sh + wv * 16 * 64;
  const int nw = gridDim.x * WPB;
  const int ntiles = N_ >> 4;
  const int ncol  = lane & 15;
  const int mbase = (lane >> 4) << 3;
  for (int tile = blockIdx.x * WPB + wv; tile < ntiles; tile += nw) {
    const int row0 = tile << 4;
    for (int m = 0; m < 16; ++m) {
      const int row = row0 + m;
      for (int c = lane; c < 128; c += 32) {
        float v = (c < 64) ? n[row * 64 + c] : aggr[row * 64 + (c - 64)];
        xbuf[m * 128 + c] = (bf16_t)v;
      }
    }
    float outv[4][8];
    mlp_tile<4>(xbuf, 128, hbuf, W1t, 128, b1, W2t, b2, g, be, lane, outv);
#pragma unroll
    for (int r = 0; r < 8; ++r) {
      const int row = row0 + mbase + r;
#pragma unroll
      for (int nt = 0; nt < 4; ++nt) {
        const int N = nt * 16 + ncol;
        n[row * 64 + N] = n[row * 64 + N] + outv[nt][r];
      }
    }
  }
}

// ---------------- decoder: out = lrelu(n@W1+b1) @ W2 + b2  (W2 padded to 16 cols) ------
__global__ void __launch_bounds__(256) decoder_kernel(const float* __restrict__ n,
    const bf16_t* __restrict__ W1t, const float* __restrict__ b1,
    const bf16_t* __restrict__ W2t, const float* __restrict__ b2,
    float* __restrict__ out, int N_) {
  __shared__ alignas(16) bf16_t sx[WPB * 16 * 64];
  __shared__ alignas(16) bf16_t sh[WPB * 16 * 64];
  const int lane = threadIdx.x & 31;
  const int wv   = threadIdx.x >> 5;
  bf16_t* xbuf = sx + wv * 16 * 64;
  bf16_t* hbuf = sh + wv * 16 * 64;
  const int nw = gridDim.x * WPB;
  const int ntiles = N_ >> 4;
  const int ncol  = lane & 15;
  const int mbase = (lane >> 4) << 3;
  for (int tile = blockIdx.x * WPB + wv; tile < ntiles; tile += nw) {
    const int row0 = tile << 4;
    for (int m = 0; m < 16; ++m)
      for (int c = lane; c < 64; c += 32)
        xbuf[m * 64 + c] = (bf16_t)n[(row0 + m) * 64 + c];

    v8f acc[4] = {zero8(), zero8(), zero8(), zero8()};
#pragma unroll 1
    for (int kc = 0; kc < 2; ++kc) {
      v16bf a = load_frag(xbuf + kc * 32, 64, lane);
#pragma unroll
      for (int nt = 0; nt < 4; ++nt) {
        v16bf b = load_frag(W1t + nt * 16 * 64 + kc * 32, 64, lane);
        acc[nt] = wmma_bf16(a, b, acc[nt]);
      }
    }
#pragma unroll
    for (int nt = 0; nt < 4; ++nt) {
      const int N = nt * 16 + ncol;
      const float bb = b1[N];
#pragma unroll
      for (int r = 0; r < 8; ++r)
        hbuf[(mbase + r) * 64 + N] = (bf16_t)lrelu(acc[nt][r] + bb);
    }
    v8f acc2 = zero8();
#pragma unroll 1
    for (int kc = 0; kc < 2; ++kc) {
      v16bf a = load_frag(hbuf + kc * 32, 64, lane);
      v16bf b = load_frag(W2t + kc * 32, 64, lane);
      acc2 = wmma_bf16(a, b, acc2);
    }
    if (ncol < 3) {
      const float bb = b2[ncol];
#pragma unroll
      for (int r = 0; r < 8; ++r)
        out[(row0 + mbase + r) * 3 + ncol] = acc2[r] + bb;
    }
  }
}

// ---------------------------------------------------------------------------------------
extern "C" void kernel_launch(void* const* d_in, const int* in_sizes, int n_in,
                              void* d_out, int out_size, void* d_ws, size_t ws_size,
                              hipStream_t stream) {
  (void)in_sizes; (void)n_in; (void)out_size; (void)ws_size;
  const int Nn = 50000, Ee = 300000;

  const float* nodes    = (const float*)d_in[0];
  const float* edges    = (const float*)d_in[1];
  const int*   edge_idx = (const int*)d_in[2];
  const float* ne_W1 = (const float*)d_in[3];
  const float* ne_b1 = (const float*)d_in[4];
  const float* ne_W2 = (const float*)d_in[5];
  const float* ne_b2 = (const float*)d_in[6];
  const float* ne_g  = (const float*)d_in[7];
  const float* ne_be = (const float*)d_in[8];
  const float* ee_W1 = (const float*)d_in[9];
  const float* ee_b1 = (const float*)d_in[10];
  const float* ee_W2 = (const float*)d_in[11];
  const float* ee_b2 = (const float*)d_in[12];
  const float* ee_g  = (const float*)d_in[13];
  const float* ee_be = (const float*)d_in[14];
  const float* nm_W1 = (const float*)d_in[15];
  const float* nm_b1 = (const float*)d_in[16];
  const float* nm_W2 = (const float*)d_in[17];
  const float* nm_b2 = (const float*)d_in[18];
  const float* nm_g  = (const float*)d_in[19];
  const float* nm_be = (const float*)d_in[20];
  const float* em_W1 = (const float*)d_in[21];
  const float* em_b1 = (const float*)d_in[22];
  const float* em_W2 = (const float*)d_in[23];
  const float* em_b2 = (const float*)d_in[24];
  const float* em_g  = (const float*)d_in[25];
  const float* em_be = (const float*)d_in[26];
  const float* dec_W1 = (const float*)d_in[27];
  const float* dec_b1 = (const float*)d_in[28];
  const float* dec_W2 = (const float*)d_in[29];
  const float* dec_b2 = (const float*)d_in[30];

  char* ws = (char*)d_ws;
  size_t off = 0;
  auto salloc = [&](size_t bytes) -> void* {
    void* p = ws + off;
    off += (bytes + 255) & ~(size_t)255;
    return p;
  };
  float* nbuf = (float*)salloc((size_t)Nn * 64 * 4);
  float* ebuf = (float*)salloc((size_t)Ee * 64 * 4);
  float* aggr = (float*)salloc((size_t)Nn * 64 * 4);
  bf16_t* neW1t  = (bf16_t*)salloc(64 * 32 * 2);
  bf16_t* neW2t  = (bf16_t*)salloc(64 * 64 * 2);
  bf16_t* eeW1t  = (bf16_t*)salloc(64 * 32 * 2);
  bf16_t* eeW2t  = (bf16_t*)salloc(64 * 64 * 2);
  bf16_t* nmW1t  = (bf16_t*)salloc(3 * 64 * 128 * 2);
  bf16_t* nmW2t  = (bf16_t*)salloc(3 * 64 * 64 * 2);
  bf16_t* emW1t  = (bf16_t*)salloc(3 * 64 * 192 * 2);
  bf16_t* emW2t  = (bf16_t*)salloc(3 * 64 * 64 * 2);
  bf16_t* decW1t = (bf16_t*)salloc(64 * 64 * 2);
  bf16_t* decW2t = (bf16_t*)salloc(16 * 64 * 2);

  auto tr = [&](const float* W, bf16_t* Wt, int K, int Nc, int Kp, int Np) {
    int total = Kp * Np;
    transpose_cvt_kernel<<<(total + 255) / 256, 256, 0, stream>>>(W, Wt, K, Nc, Kp, Np);
  };
  tr(ne_W1, neW1t, 16, 64, 32, 64);
  tr(ne_W2, neW2t, 64, 64, 64, 64);
  tr(ee_W1, eeW1t, 8, 64, 32, 64);
  tr(ee_W2, eeW2t, 64, 64, 64, 64);
  for (int i = 0; i < 3; ++i) {
    tr(nm_W1 + i * 128 * 64, nmW1t + i * 64 * 128, 128, 64, 128, 64);
    tr(nm_W2 + i * 64 * 64,  nmW2t + i * 64 * 64,  64, 64, 64, 64);
    tr(em_W1 + i * 192 * 64, emW1t + i * 64 * 192, 192, 64, 192, 64);
    tr(em_W2 + i * 64 * 64,  emW2t + i * 64 * 64,  64, 64, 64, 64);
  }
  tr(dec_W1, decW1t, 64, 64, 64, 64);
  tr(dec_W2, decW2t, 64, 3, 64, 16);

  const int ntileN = Nn / 16, ntileE = Ee / 16;
  const int blkN = (ntileN + WPB - 1) / WPB;
  const int blkE = (ntileE + WPB - 1) / WPB;

  encoder_kernel<16><<<blkN, 256, 0, stream>>>(nodes, neW1t, ne_b1, neW2t, ne_b2,
                                               ne_g, ne_be, nbuf, Nn);
  encoder_kernel<8><<<blkE, 256, 0, stream>>>(edges, eeW1t, ee_b1, eeW2t, ee_b2,
                                              ee_g, ee_be, ebuf, Ee);

  for (int i = 0; i < 3; ++i) {
    int cnt = Nn * 64;
    zero_kernel<<<(cnt + 255) / 256, 256, 0, stream>>>(aggr, cnt);
    edge_update_kernel<<<blkE, 256, 0, stream>>>(ebuf, nbuf, edge_idx,
        emW1t + i * 64 * 192, em_b1 + i * 64, emW2t + i * 64 * 64, em_b2 + i * 64,
        em_g + i * 64, em_be + i * 64, aggr, Ee);
    node_update_kernel<<<blkN, 256, 0, stream>>>(nbuf, aggr,
        nmW1t + i * 64 * 128, nm_b1 + i * 64, nmW2t + i * 64 * 64, nm_b2 + i * 64,
        nm_g + i * 64, nm_be + i * 64, Nn);
  }
  decoder_kernel<<<blkN, 256, 0, stream>>>(nbuf, decW1t, dec_b1, decW2t, dec_b2,
                                           (float*)d_out, Nn);
}
